// HeadAttention_11424613007435
// MI455X (gfx1250) — matrix operand order repaired and checked by reading, not compiled, not run on hardware
//
#include <hip/hip_runtime.h>

typedef _Float16 v16h  __attribute__((ext_vector_type(16)));
typedef _Float16 half8 __attribute__((ext_vector_type(8)));
typedef _Float16 half4 __attribute__((ext_vector_type(4)));
typedef float    v8f   __attribute__((ext_vector_type(8)));

#define BB 4
#define SS 4096
#define EE 2048
#define HH 128

static __device__ __forceinline__ v16h mk16(half8 a, half8 b) {
  v16h r;
#pragma unroll
  for (int i = 0; i < 8; ++i) { r[i] = a[i]; r[8 + i] = b[i]; }
  return r;
}

// ---------------------------------------------------------------------------
// Weight pre-conversion: (E,H) fp32 -> transposed (H,E) f16
// ---------------------------------------------------------------------------
__global__ __launch_bounds__(256) void wconv_kernel(
    const float* __restrict__ W, _Float16* __restrict__ Wt) {
  const int idx = blockIdx.x * 256 + threadIdx.x;   // over E*H, coalesced read
  const int e = idx >> 7;          // / HH
  const int h = idx & (HH - 1);
  Wt[(size_t)h * EE + e] = (_Float16)W[idx];
}

// ---------------------------------------------------------------------------
// Stage 1 (fused): x (B*S,E) fp32 read ONCE -> Q,K,V (B*S,H) f16
// grid.x = (B*S)/128, block = 512 (16 waves)
// wave w: rows 16*(w>>1) .. +15, cols 64*(w&1) .. +63 (4 n-tiles)
// ---------------------------------------------------------------------------
__global__ __launch_bounds__(512) void qkv_fused_kernel(
    const float* __restrict__ x,
    const _Float16* __restrict__ Wt0, const _Float16* __restrict__ Wt1,
    const _Float16* __restrict__ Wt2,
    const float* __restrict__ b0, const float* __restrict__ b1,
    const float* __restrict__ b2,
    _Float16* __restrict__ Qh, _Float16* __restrict__ Kh,
    _Float16* __restrict__ Vh, float qscale)
{
  __shared__ __align__(16) _Float16 xs[128][40];   // x tile, f16

  const int t  = threadIdx.x;
  const int w  = t >> 5;
  const int l  = t & 31;
  const int lh = l >> 4;
  const int ln = l & 15;
  const int r0 = blockIdx.x * 128;
  const int rowg = (w >> 1) * 16;    // wave's row group
  const int col0 = (w & 1) * 64;     // wave's column half

  const _Float16* Wts[3] = {Wt0, Wt1, Wt2};

  const v8f vz = {0.f,0.f,0.f,0.f,0.f,0.f,0.f,0.f};
  v8f acc[3][4];
#pragma unroll
  for (int m = 0; m < 3; ++m)
#pragma unroll
    for (int nt = 0; nt < 4; ++nt) acc[m][nt] = vz;

  for (int kc = 0; kc < EE; kc += 32) {
    __syncthreads();
    { // x tile 128x32 fp32 -> f16 LDS, packed b64 stores
      const int rowIn = t >> 3, seg = t & 7;
#pragma unroll
      for (int pass = 0; pass < 2; ++pass) {
        const int row = pass * 64 + rowIn;
        const float4 v = *(const float4*)&x[(size_t)(r0 + row) * EE + kc + seg * 4];
        const half4 hv = {(_Float16)v.x, (_Float16)v.y, (_Float16)v.z, (_Float16)v.w};
        *(half4*)&xs[row][seg * 4] = hv;
      }
    }
    if (kc + 32 < EE && t < 128)     // prefetch next x chunk (one line per row)
      __builtin_prefetch(&x[(size_t)(r0 + t) * EE + kc + 32], 0, 0);
    __syncthreads();

    // A fragment: 16x32 f16 (ISA A-layout), shared across Q/K/V
    v16h a;
    {
      const int row = rowg + ln;
#pragma unroll
      for (int v = 0; v < 8; ++v) {
        const int kb = ((v & 4) ? 16 : 0) + lh * 8 + 2 * (v & 3);
        a[2*v]   = xs[row][kb];
        a[2*v+1] = xs[row][kb+1];
      }
    }
#pragma unroll
    for (int m = 0; m < 3; ++m) {
#pragma unroll
      for (int nt = 0; nt < 4; ++nt) {
        const int h = col0 + nt * 16 + ln;
        const half8* wp = (const half8*)&Wts[m][(size_t)h * EE + kc + lh * 8];
        const v16h bfr = mk16(wp[0], wp[2]);
        acc[m][nt] = __builtin_amdgcn_wmma_f32_16x16x32_f16(
            false, a, false, bfr, (short)0, acc[m][nt], false, false);
      }
    }
  }

  // bias + scale + f16 stores
  const float* biasp[3] = {b0, b1, b2};
  _Float16* outp[3] = {Qh, Kh, Vh};
#pragma unroll
  for (int m = 0; m < 3; ++m) {
    const float sc = (m == 0) ? qscale : 1.0f;
#pragma unroll
    for (int nt = 0; nt < 4; ++nt) {
      const int h = col0 + nt * 16 + ln;
      const float bv = biasp[m][h];
#pragma unroll
      for (int r = 0; r < 8; ++r) {
        const int row = r0 + rowg + r + 8 * lh;
        outp[m][(size_t)row * HH + h] = (_Float16)((acc[m][nt][r] + bv) * sc);
      }
    }
  }
}

// ---------------------------------------------------------------------------
// Stage 2: causal flash attention over f16 Q,K,V -> fp32 out
// grid = (S/128, B), block = 256; wave w owns query rows qw..qw+15
// ---------------------------------------------------------------------------
__global__ __launch_bounds__(256) void attn_kernel(
    const _Float16* __restrict__ Qh, const _Float16* __restrict__ Kh,
    const _Float16* __restrict__ Vh, float* __restrict__ out)
{
  __shared__ __align__(16) _Float16 Ks[64][136];    // K tile row-major [key][h]
  __shared__ __align__(16) _Float16 Vts[128][72];   // V tile transposed [h][key]
  __shared__ __align__(16) _Float16 Ps[8][16][72];  // per-wave P scratch

  const int t  = threadIdx.x;
  const int w  = t >> 5;
  const int l  = t & 31;
  const int lh = l >> 4;
  const int ln = l & 15;
  const int b  = blockIdx.y;
  const int q0 = blockIdx.x * 128;
  const int qw = q0 + w * 16;
  const size_t base = (size_t)b * SS * HH;

  // Q fragments (4 chunks of K-dim 32 over H=128), loaded once
  v16h qf[4];
  {
    const _Float16* qp = Qh + base + (size_t)(qw + ln) * HH;
#pragma unroll
    for (int hc = 0; hc < 4; ++hc) {
      const half8 a0 = *(const half8*)&qp[hc * 32 + lh * 8];
      const half8 a1 = *(const half8*)&qp[hc * 32 + lh * 8 + 16];
      qf[hc] = mk16(a0, a1);
    }
  }

  const v8f vz = {0.f,0.f,0.f,0.f,0.f,0.f,0.f,0.f};
  v8f o[8];
#pragma unroll
  for (int i = 0; i < 8; ++i) o[i] = vz;
  float mi[8], li[8];
#pragma unroll
  for (int r = 0; r < 8; ++r) { mi[r] = -1e30f; li[r] = 0.f; }

  const int jmax = (q0 + 127) >> 6;
  for (int j = 0; j <= jmax; ++j) {
    const int kb = j * 64;
    __syncthreads();
    // K tile: row-major, b128 both sides
#pragma unroll
    for (int pass = 0; pass < 4; ++pass) {
      const int idx = pass * 256 + t;
      const int row = idx >> 4, seg = idx & 15;
      const half8 kv = *(const half8*)&Kh[base + (size_t)(kb + row) * HH + seg * 8];
      *(half8*)&Ks[row][seg * 8] = kv;
    }
    // V tile: 8x4 register transpose per thread, b128 ds stores
    {
      const int kblk = t >> 5;     // 8 keys
      const int hseg = t & 31;     // 4 h
      half4 vv[8];
#pragma unroll
      for (int i = 0; i < 8; ++i)
        vv[i] = *(const half4*)&Vh[base + (size_t)(kb + kblk * 8 + i) * HH + hseg * 4];
#pragma unroll
      for (int c = 0; c < 4; ++c) {
        half8 rw;
#pragma unroll
        for (int i = 0; i < 8; ++i) rw[i] = vv[i][c];
        *(half8*)&Vts[hseg * 4 + c][kblk * 8] = rw;
      }
    }
    __syncthreads();

    if (kb <= qw + 15) {      // wave-uniform: EXEC stays all-ones for WMMA
      // scores S = Q K^T : 4 key sub-tiles x 4 h-chunks
      v8f sc[4];
#pragma unroll
      for (int kt = 0; kt < 4; ++kt) {
        sc[kt] = vz;
#pragma unroll
        for (int hc = 0; hc < 4; ++hc) {
          const half8* p = (const half8*)&Ks[kt * 16 + ln][hc * 32 + lh * 8];
          const v16h bf = mk16(p[0], p[2]);
          sc[kt] = __builtin_amdgcn_wmma_f32_16x16x32_f16(
              false, qf[hc], false, bf, (short)0, sc[kt], false, false);
        }
      }
      // causal mask on diagonal-overlapping tiles
      if (kb + 63 > qw) {
#pragma unroll
        for (int kt = 0; kt < 4; ++kt)
#pragma unroll
          for (int r = 0; r < 8; ++r) {
            const int qg = qw + r + 8 * lh;
            const int kg = kb + kt * 16 + ln;
            if (kg > qg) sc[kt][r] = -1e30f;
          }
      }
      // online softmax (row reductions across the 16-lane half)
#pragma unroll
      for (int r = 0; r < 8; ++r) {
        float m0 = fmaxf(fmaxf(sc[0][r], sc[1][r]), fmaxf(sc[2][r], sc[3][r]));
#pragma unroll
        for (int off = 1; off < 16; off <<= 1)
          m0 = fmaxf(m0, __shfl_xor(m0, off, 32));
        const float mn = fmaxf(mi[r], m0);
        const float alpha = __expf(mi[r] - mn);
        float rs = 0.f;
#pragma unroll
        for (int kt = 0; kt < 4; ++kt) {
          const float p = __expf(sc[kt][r] - mn);
          rs += p;
          Ps[w][r + 8 * lh][kt * 16 + ln] = (_Float16)p;
        }
#pragma unroll
        for (int off = 1; off < 16; off <<= 1)
          rs += __shfl_xor(rs, off, 32);
        li[r] = li[r] * alpha + rs;
        mi[r] = mn;
#pragma unroll
        for (int nt = 0; nt < 8; ++nt) o[nt][r] *= alpha;
      }
      asm volatile("s_wait_dscnt 0" ::: "memory");  // Ps write->read, same wave

      // P (A-layout, f16) and O += P V
      v16h pa[2];
#pragma unroll
      for (int kk = 0; kk < 2; ++kk) {
        const half8 a0 = *(const half8*)&Ps[w][ln][kk * 32 + lh * 8];
        const half8 a1 = *(const half8*)&Ps[w][ln][kk * 32 + 16 + lh * 8];
        pa[kk] = mk16(a0, a1);
      }
#pragma unroll
      for (int nt = 0; nt < 8; ++nt)
#pragma unroll
        for (int kk = 0; kk < 2; ++kk) {
          const half8* vp = (const half8*)&Vts[nt * 16 + ln][kk * 32 + lh * 8];
          const v16h vf = mk16(vp[0], vp[2]);
          o[nt] = __builtin_amdgcn_wmma_f32_16x16x32_f16(
              false, pa[kk], false, vf, (short)0, o[nt], false, false);
        }
    }
  }

  // epilogue: O / l
#pragma unroll
  for (int r = 0; r < 8; ++r) {
    const float inv = 1.0f / li[r];
    const int qg = qw + r + 8 * lh;
#pragma unroll
    for (int nt = 0; nt < 8; ++nt)
      out[base + (size_t)qg * HH + nt * 16 + ln] = o[nt][r] * inv;
  }
}

// ---------------------------------------------------------------------------
extern "C" void kernel_launch(void* const* d_in, const int* in_sizes, int n_in,
                              void* d_out, int out_size, void* d_ws, size_t ws_size,
                              hipStream_t stream) {
  const float* x  = (const float*)d_in[0];
  const float* Wq = (const float*)d_in[1];
  const float* bq = (const float*)d_in[2];
  const float* Wk = (const float*)d_in[3];
  const float* bk = (const float*)d_in[4];
  const float* Wv = (const float*)d_in[5];
  const float* bv = (const float*)d_in[6];

  const size_t nQKV = (size_t)BB * SS * HH;
  const size_t nW   = (size_t)EE * HH;
  _Float16* Qh  = (_Float16*)d_ws;
  _Float16* Kh  = Qh + nQKV;
  _Float16* Vh  = Kh + nQKV;
  _Float16* Wtq = Vh + nQKV;
  _Float16* Wtk = Wtq + nW;
  _Float16* Wtv = Wtk + nW;

  const float qscale = 0.08838834764831845f;  // 1/sqrt(H)

  wconv_kernel<<<dim3(nW / 256), 256, 0, stream>>>(Wq, Wtq);
  wconv_kernel<<<dim3(nW / 256), 256, 0, stream>>>(Wk, Wtk);
  wconv_kernel<<<dim3(nW / 256), 256, 0, stream>>>(Wv, Wtv);

  qkv_fused_kernel<<<dim3((BB * SS) / 128), 512, 0, stream>>>(
      x, Wtq, Wtk, Wtv, bq, bk, bv, Qh, Kh, Vh, qscale);

  attn_kernel<<<dim3(SS / 128, BB), 256, 0, stream>>>(Qh, Kh, Vh, (float*)d_out);
}
